// Int8WeightOnlyLinear_39152921870844
// MI455X (gfx1250) — matrix operand order repaired and checked by reading, not compiled, not run on hardware
//
#include <hip/hip_runtime.h>

// int8 weight-only linear as f16x2-split WMMA GEMM for gfx1250 (MI455X).
// out[m,n] = scale[n] * sum_k x[m,k]*w_int8[n,k] + bias[n]
// M = 8192 (4*2048), N = 8192, K = 2048.
//
// Fast path (needs 96MB workspace): prep kernels split x into f16 hi+lo
// (exact to ~2^-22) and convert int8 weights to f16 (exact) ONCE; the GEMM
// loop is then async-copy -> LDS -> WMMA with ~zero VALU. Staging uses
// GLOBAL_LOAD_ASYNC_TO_LDS_B128 (ASYNCcnt) when the toolchain exposes it,
// overlapping the tile-(i+1) DMA with tile-i WMMA compute.
// Fallback path: fused convert-in-loop version.
//
// GEMM: block tile 128(M) x 256(N), K-step 64, 8 waves (wave32),
// wave tile 64x64, double-buffered LDS (2 x 72KB of the WGP's 320KB).

typedef __attribute__((ext_vector_type(16))) _Float16 v16h;
typedef __attribute__((ext_vector_type(8)))  _Float16 v8h;
typedef __attribute__((ext_vector_type(8)))  float    v8f;
typedef __attribute__((ext_vector_type(4)))  float    v4f;
typedef __attribute__((ext_vector_type(4)))  int      v4i;

#define K_DIM 2048
#define N_DIM 8192
#define M_DIM 8192
#define BLK_M 128
#define BLK_N 256
#define BLK_K 64
#define LDSS  72   // LDS row stride in f16 units: 64 + 8 pad (bank-conflict-free)

#define AH_OFF 0
#define AL_OFF (128 * LDSS)
#define BF_OFF (256 * LDSS)

#if defined(__has_builtin)
#if __has_builtin(__builtin_amdgcn_global_load_async_to_lds_b128)
#define HAVE_ASYNC_LDS 1
#endif
#endif

#ifdef HAVE_ASYNC_LDS
// Builtin signature (from clang diagnostic): (as1 v4i*, as3 v4i*, Imm int, Imm int)
typedef __attribute__((address_space(1))) v4i g_v4i;
typedef __attribute__((address_space(3))) v4i l_v4i;
#define ASYNC_B128(GS, LS, OFF) \
    __builtin_amdgcn_global_load_async_to_lds_b128((g_v4i*)(GS), (l_v4i*)(LS), (OFF), 0)

static __device__ __forceinline__ void wait_async0() {
#if __has_builtin(__builtin_amdgcn_s_wait_asynccnt)
    __builtin_amdgcn_s_wait_asynccnt(0);
#else
    asm volatile("s_wait_asynccnt 0x0" ::: "memory");
#endif
}

// 16 B128 async copies: A_hi 64B, A_lo 64B, B 128B per thread.
// IOFFSET is added to BOTH global and LDS addresses (strides match).
static __device__ __forceinline__ void issue_async_tile(
    const _Float16* gA, const _Float16* gL, const _Float16* gB,
    _Float16* lA, _Float16* lL, _Float16* lB)
{
    ASYNC_B128(gA, lA, 0);   ASYNC_B128(gA, lA, 16);
    ASYNC_B128(gA, lA, 32);  ASYNC_B128(gA, lA, 48);
    ASYNC_B128(gL, lL, 0);   ASYNC_B128(gL, lL, 16);
    ASYNC_B128(gL, lL, 32);  ASYNC_B128(gL, lL, 48);
    ASYNC_B128(gB, lB, 0);   ASYNC_B128(gB, lB, 16);
    ASYNC_B128(gB, lB, 32);  ASYNC_B128(gB, lB, 48);
    ASYNC_B128(gB, lB, 64);  ASYNC_B128(gB, lB, 80);
    ASYNC_B128(gB, lB, 96);  ASYNC_B128(gB, lB, 112);
}
#endif

// ======================= preprocessing kernels =========================

__global__ __launch_bounds__(256)
void prep_x_split(const float* __restrict__ x,
                  _Float16* __restrict__ xh,
                  _Float16* __restrict__ xl)
{
    const unsigned i = (blockIdx.x * 256u + threadIdx.x) * 8u;
    v4f a = *(const v4f*)(x + i);
    v4f b = *(const v4f*)(x + i + 4);
    v8h h, l;
#pragma unroll
    for (int j = 0; j < 4; ++j) {
        _Float16 ha = (_Float16)a[j];
        h[j]     = ha;
        l[j]     = (_Float16)(a[j] - (float)ha);
        _Float16 hb = (_Float16)b[j];
        h[4 + j] = hb;
        l[4 + j] = (_Float16)(b[j] - (float)hb);
    }
    *(v8h*)(xh + i) = h;
    *(v8h*)(xl + i) = l;
}

__global__ __launch_bounds__(256)
void prep_w_f16(const signed char* __restrict__ w,
                _Float16* __restrict__ wh)
{
    const unsigned i = (blockIdx.x * 256u + threadIdx.x) * 16u;
    v4i q = *(const v4i*)(w + i);
    v8h b0, b1;
#pragma unroll
    for (int j = 0; j < 16; ++j) {
        _Float16 v = (_Float16)(signed char)(q[j >> 2] >> ((j & 3) * 8));
        if (j < 8) b0[j] = v; else b1[j - 8] = v;
    }
    *(v8h*)(wh + i)     = b0;
    *(v8h*)(wh + i + 8) = b1;
}

// ======================= fast GEMM (pre-converted f16) =================

__device__ __forceinline__ void fetch_f16(const _Float16* __restrict__ xh,
                                          const _Float16* __restrict__ xl,
                                          const _Float16* __restrict__ wh,
                                          unsigned a_base, unsigned w_base, int k0,
                                          v8h (&ar)[4], v8h (&lr)[4], v8h (&br)[8])
{
#pragma unroll
    for (int i = 0; i < 4; ++i) ar[i] = *(const v8h*)(xh + a_base + k0 + i * 8);
#pragma unroll
    for (int i = 0; i < 4; ++i) lr[i] = *(const v8h*)(xl + a_base + k0 + i * 8);
#pragma unroll
    for (int i = 0; i < 8; ++i) br[i] = *(const v8h*)(wh + w_base + k0 + i * 8);
}

__global__ __launch_bounds__(256)
void int8wo_gemm_fast(const _Float16* __restrict__ xh,
                      const _Float16* __restrict__ xl,
                      const _Float16* __restrict__ wh,
                      const float* __restrict__ scale,
                      const float* __restrict__ bias,
                      float* __restrict__ out)
{
    __shared__ _Float16 smem[2][512 * LDSS];

    const int tid   = threadIdx.x;
    const int lane  = tid & 31;
    const int wave  = tid >> 5;
    const int l16   = lane & 15;
    const int lhalf = lane >> 4;
    const int wm    = wave & 1;
    const int wn    = wave >> 1;

    const int blockM = blockIdx.y * BLK_M;
    const int blockN = blockIdx.x * BLK_N;

    // staging: thread t stages A_hi/A_lo row t/2 (32 f16 at (t&1)*32)
    //          and B row t (64 f16)
    const int arow = tid >> 1;
    const int akh  = (tid & 1) * 32;
    const unsigned a_base = (unsigned)(blockM + arow) * K_DIM + akh;
    const unsigned w_base = (unsigned)(blockN + tid) * K_DIM;

    v8f acc[4][4] = {};
    int buf = 0;

#ifdef HAVE_ASYNC_LDS
    issue_async_tile(xh + a_base, xl + a_base, wh + w_base,
                     &smem[0][AH_OFF] + arow * LDSS + akh,
                     &smem[0][AL_OFF] + arow * LDSS + akh,
                     &smem[0][BF_OFF] + tid * LDSS);
#else
    v8h ar[4], lr[4], br[8];
    fetch_f16(xh, xl, wh, a_base, w_base, 0, ar, lr, br);
#endif

    for (int k0 = 0; k0 < K_DIM; k0 += BLK_K) {
        _Float16* AH = &smem[buf][AH_OFF];
        _Float16* AL = &smem[buf][AL_OFF];
        _Float16* BF = &smem[buf][BF_OFF];

#ifdef HAVE_ASYNC_LDS
        wait_async0();       // this wave's copies of tile k0 are in LDS
        __syncthreads();     // + everyone else's (also drains DScnt reads)
        if (k0 + BLK_K < K_DIM) {
            // DMA next tile into the other buffer DURING this tile's WMMAs
            _Float16* nAH = &smem[buf ^ 1][AH_OFF];
            _Float16* nAL = &smem[buf ^ 1][AL_OFF];
            _Float16* nBF = &smem[buf ^ 1][BF_OFF];
            issue_async_tile(xh + a_base + k0 + BLK_K,
                             xl + a_base + k0 + BLK_K,
                             wh + w_base + k0 + BLK_K,
                             nAH + arow * LDSS + akh,
                             nAL + arow * LDSS + akh,
                             nBF + tid * LDSS);
        }
#else
        {
            _Float16* ah = AH + arow * LDSS + akh;
            _Float16* al = AL + arow * LDSS + akh;
#pragma unroll
            for (int i = 0; i < 4; ++i) {
                *(v8h*)(ah + i * 8) = ar[i];
                *(v8h*)(al + i * 8) = lr[i];
            }
            _Float16* bp = BF + tid * LDSS;
#pragma unroll
            for (int i = 0; i < 8; ++i)
                *(v8h*)(bp + i * 8) = br[i];
        }
        if (k0 + BLK_K < K_DIM)
            fetch_f16(xh, xl, wh, a_base, w_base, k0 + BLK_K, ar, lr, br);
        __syncthreads();
#endif

        // 2 k-halves x 4x4 subtiles x (hi+lo) = 64 WMMAs per k-step
#pragma unroll
        for (int kk = 0; kk < 2; ++kk) {
            v16h bfr[4];
#pragma unroll
            for (int nt = 0; nt < 4; ++nt) {
                const _Float16* p = BF + (wn * 64 + nt * 16 + l16) * LDSS
                                       + kk * 32 + lhalf * 8;
                v8h b0 = *(const v8h*)p;
                v8h b1 = *(const v8h*)(p + 16);
                bfr[nt] = __builtin_shufflevector(b0, b1,
                    0,1,2,3,4,5,6,7,8,9,10,11,12,13,14,15);
            }
#pragma unroll
            for (int mt = 0; mt < 4; ++mt) {
                const int rofs = (wm * 64 + mt * 16 + l16) * LDSS + kk * 32 + lhalf * 8;
                v8h h0 = *(const v8h*)(AH + rofs);
                v8h h1 = *(const v8h*)(AH + rofs + 16);
                v8h l0 = *(const v8h*)(AL + rofs);
                v8h l1 = *(const v8h*)(AL + rofs + 16);
                v16h ahf = __builtin_shufflevector(h0, h1,
                    0,1,2,3,4,5,6,7,8,9,10,11,12,13,14,15);
                v16h alf = __builtin_shufflevector(l0, l1,
                    0,1,2,3,4,5,6,7,8,9,10,11,12,13,14,15);
#pragma unroll
                for (int nt = 0; nt < 4; ++nt) {
                    acc[mt][nt] = __builtin_amdgcn_wmma_f32_16x16x32_f16(
                        false, ahf, false, bfr[nt], (short)0, acc[mt][nt], false, false);
                    acc[mt][nt] = __builtin_amdgcn_wmma_f32_16x16x32_f16(
                        false, alf, false, bfr[nt], (short)0, acc[mt][nt], false, false);
                }
            }
        }
        buf ^= 1;
    }

#pragma unroll
    for (int nt = 0; nt < 4; ++nt) {
        const int n = blockN + wn * 64 + nt * 16 + l16;
        const float s  = scale[n];
        const float bv = bias[n];
#pragma unroll
        for (int mt = 0; mt < 4; ++mt) {
            const int mb = blockM + wm * 64 + mt * 16 + 8 * lhalf;
#pragma unroll
            for (int r = 0; r < 8; ++r) {
                __builtin_nontemporal_store(acc[mt][nt][r] * s + bv,
                                            out + (size_t)(mb + r) * N_DIM + n);
            }
        }
    }
}

// ======================= fused fallback (no workspace) =================

__device__ __forceinline__ void fetch_raw(const float* __restrict__ x,
                                          const signed char* __restrict__ w,
                                          unsigned x_base, unsigned w_base, int k0,
                                          v4f (&xr)[8], v4i (&wr)[4])
{
#pragma unroll
    for (int i = 0; i < 8; ++i) xr[i] = *(const v4f*)(x + x_base + k0 + i * 4);
#pragma unroll
    for (int i = 0; i < 4; ++i) wr[i] = *(const v4i*)(w + w_base + k0 + i * 16);
}

__global__ __launch_bounds__(256)
void int8wo_gemm_fused(const float* __restrict__ x,
                       const signed char* __restrict__ w,
                       const float* __restrict__ scale,
                       const float* __restrict__ bias,
                       float* __restrict__ out)
{
    __shared__ _Float16 smem[2][512 * LDSS];

    const int tid   = threadIdx.x;
    const int lane  = tid & 31;
    const int wave  = tid >> 5;
    const int l16   = lane & 15;
    const int lhalf = lane >> 4;
    const int wm    = wave & 1;
    const int wn    = wave >> 1;

    const int blockM = blockIdx.y * BLK_M;
    const int blockN = blockIdx.x * BLK_N;

    const int arow = tid >> 1;
    const int akh  = (tid & 1) * 32;
    const unsigned x_base = (unsigned)(blockM + arow) * K_DIM + akh;
    const unsigned w_base = (unsigned)(blockN + tid) * K_DIM;

    v8f acc[4][4] = {};

    v4f xr[8];
    v4i wr[4];
    fetch_raw(x, w, x_base, w_base, 0, xr, wr);

    int buf = 0;
    for (int k0 = 0; k0 < K_DIM; k0 += BLK_K) {
        _Float16* AH = &smem[buf][AH_OFF];
        _Float16* AL = &smem[buf][AL_OFF];
        _Float16* BF = &smem[buf][BF_OFF];

        {
            v8h hv[4], lv[4];
#pragma unroll
            for (int c = 0; c < 8; ++c)
#pragma unroll
                for (int j = 0; j < 4; ++j) {
                    float f = xr[c][j];
                    _Float16 h = (_Float16)f;
                    hv[c >> 1][(c & 1) * 4 + j] = h;
                    lv[c >> 1][(c & 1) * 4 + j] = (_Float16)(f - (float)h);
                }
            _Float16* ah = AH + arow * LDSS + akh;
            _Float16* al = AL + arow * LDSS + akh;
#pragma unroll
            for (int c = 0; c < 4; ++c) {
                *(v8h*)(ah + c * 8) = hv[c];
                *(v8h*)(al + c * 8) = lv[c];
            }
            _Float16* bp = BF + tid * LDSS;
#pragma unroll
            for (int c = 0; c < 4; ++c) {
                v8h b0, b1;
#pragma unroll
                for (int j = 0; j < 16; ++j) {
                    int word = wr[c][j >> 2];
                    _Float16 v = (_Float16)(signed char)(word >> ((j & 3) * 8));
                    if (j < 8) b0[j] = v; else b1[j - 8] = v;
                }
                *(v8h*)(bp + c * 16)     = b0;
                *(v8h*)(bp + c * 16 + 8) = b1;
            }
        }

        if (k0 + BLK_K < K_DIM)
            fetch_raw(x, w, x_base, w_base, k0 + BLK_K, xr, wr);

        __syncthreads();

#pragma unroll
        for (int kk = 0; kk < 2; ++kk) {
            v16h bfr[4];
#pragma unroll
            for (int nt = 0; nt < 4; ++nt) {
                const _Float16* p = BF + (wn * 64 + nt * 16 + l16) * LDSS
                                       + kk * 32 + lhalf * 8;
                v8h b0 = *(const v8h*)p;
                v8h b1 = *(const v8h*)(p + 16);
                bfr[nt] = __builtin_shufflevector(b0, b1,
                    0,1,2,3,4,5,6,7,8,9,10,11,12,13,14,15);
            }
#pragma unroll
            for (int mt = 0; mt < 4; ++mt) {
                const int rofs = (wm * 64 + mt * 16 + l16) * LDSS + kk * 32 + lhalf * 8;
                v8h h0 = *(const v8h*)(AH + rofs);
                v8h h1 = *(const v8h*)(AH + rofs + 16);
                v8h l0 = *(const v8h*)(AL + rofs);
                v8h l1 = *(const v8h*)(AL + rofs + 16);
                v16h ahf = __builtin_shufflevector(h0, h1,
                    0,1,2,3,4,5,6,7,8,9,10,11,12,13,14,15);
                v16h alf = __builtin_shufflevector(l0, l1,
                    0,1,2,3,4,5,6,7,8,9,10,11,12,13,14,15);
#pragma unroll
                for (int nt = 0; nt < 4; ++nt) {
                    acc[mt][nt] = __builtin_amdgcn_wmma_f32_16x16x32_f16(
                        false, ahf, false, bfr[nt], (short)0, acc[mt][nt], false, false);
                    acc[mt][nt] = __builtin_amdgcn_wmma_f32_16x16x32_f16(
                        false, alf, false, bfr[nt], (short)0, acc[mt][nt], false, false);
                }
            }
        }
        buf ^= 1;
    }

#pragma unroll
    for (int nt = 0; nt < 4; ++nt) {
        const int n = blockN + wn * 64 + nt * 16 + l16;
        const float s  = scale[n];
        const float bv = bias[n];
#pragma unroll
        for (int mt = 0; mt < 4; ++mt) {
            const int mb = blockM + wm * 64 + mt * 16 + 8 * lhalf;
#pragma unroll
            for (int r = 0; r < 8; ++r) {
                __builtin_nontemporal_store(acc[mt][nt][r] * s + bv,
                                            out + (size_t)(mb + r) * N_DIM + n);
            }
        }
    }
}

// ======================= launch ========================================

extern "C" void kernel_launch(void* const* d_in, const int* in_sizes, int n_in,
                              void* d_out, int out_size, void* d_ws, size_t ws_size,
                              hipStream_t stream) {
    const float*       x     = (const float*)d_in[0];
    const signed char* w     = (const signed char*)d_in[1];
    const float*       scale = (const float*)d_in[2];
    const float*       bias  = (const float*)d_in[3];
    float*             out   = (float*)d_out;

    const size_t xs_elems = (size_t)M_DIM * K_DIM;      // 16.7M
    const size_t ws_elems = (size_t)N_DIM * K_DIM;      // 16.7M
    const size_t need = (2 * xs_elems + ws_elems) * sizeof(_Float16);  // 96 MB

    dim3 grid(N_DIM / BLK_N, M_DIM / BLK_M);            // 32 x 64 workgroups

    if (ws_size >= need) {
        _Float16* xh = (_Float16*)d_ws;
        _Float16* xl = xh + xs_elems;
        _Float16* wh = xl + xs_elems;
        prep_x_split<<<dim3(xs_elems / 8 / 256), dim3(256), 0, stream>>>(x, xh, xl);
        prep_w_f16 <<<dim3(ws_elems / 16 / 256), dim3(256), 0, stream>>>(w, wh);
        int8wo_gemm_fast<<<grid, dim3(256), 0, stream>>>(xh, xl, wh, scale, bias, out);
    } else {
        int8wo_gemm_fused<<<grid, dim3(256), 0, stream>>>(x, w, scale, bias, out);
    }
}
